// MSGAD_34282428956756
// MI455X (gfx1250) — matrix operand order repaired and checked by reading, not compiled, not run on hardware
//
#include <hip/hip_runtime.h>

typedef _Float16 v8h  __attribute__((ext_vector_type(8)));
typedef _Float16 v16h __attribute__((ext_vector_type(16)));
typedef float    v8f  __attribute__((ext_vector_type(8)));

#define N_NODES 8192
#define N_EDGES 262144
#define IN_DIM  256
#define H_DIM   128
#define NH      (N_NODES * H_DIM)   // 1,048,576

// ---------------------------------------------------------------------------
// Kernel 1: h = leaky_relu(X @ W + b); also init feat=h, agg=0, acc_t=theta[t][0]*h
// ---------------------------------------------------------------------------
__global__ void msgad_gemm_leaky(const float* __restrict__ X,
                                 const float* __restrict__ W,
                                 const float* __restrict__ b,
                                 float* __restrict__ feat,
                                 float* __restrict__ agg,
                                 float* __restrict__ acc0, float* __restrict__ acc1,
                                 float* __restrict__ acc2, float* __restrict__ acc3) {
    __shared__ float xs[IN_DIM];
    const int n = blockIdx.x;
    const int t = threadIdx.x;            // 0..127
    xs[t]       = X[n * IN_DIM + t];
    xs[t + 128] = X[n * IN_DIM + t + 128];
    __syncthreads();
    float sum = b[t];
#pragma unroll 8
    for (int j = 0; j < IN_DIM; ++j)
        sum = fmaf(xs[j], W[j * H_DIM + t], sum);
    const float v = sum > 0.0f ? sum : 0.01f * sum;   // leaky_relu slope 0.01
    const int idx = n * H_DIM + t;
    feat[idx] = v;
    agg[idx]  = 0.0f;
    acc0[idx] = 2.5f * v;   // theta[0][0]
    acc1[idx] = 0.0f;       // theta[1][0] = 0
    acc2[idx] = 0.0f;
    acc3[idx] = 0.0f;
}

// ---------------------------------------------------------------------------
// Degree / dinv
// ---------------------------------------------------------------------------
__global__ void msgad_deg_init(float* __restrict__ deg) {
    const int i = blockIdx.x * 256 + threadIdx.x;
    if (i < N_NODES) deg[i] = 1.0f;   // self-loop
}

__global__ void msgad_deg_count(const int* __restrict__ src, float* __restrict__ deg) {
    const int e = blockIdx.x * 256 + threadIdx.x;
    if (e < N_EDGES)
        (void)unsafeAtomicAdd(&deg[src[e]], 1.0f);   // global_atomic_add_f32
}

__global__ void msgad_dinv(const float* __restrict__ deg, float* __restrict__ dinv) {
    const int i = blockIdx.x * 256 + threadIdx.x;
    if (i < N_NODES) dinv[i] = rsqrtf(fmaxf(deg[i], 1.0f));
}

// ---------------------------------------------------------------------------
// Scatter: agg[d] += feat[s] * dinv[s]  over E edges + N self loops.
// One thread handles 4 consecutive features of one edge slot.
// ---------------------------------------------------------------------------
__global__ void msgad_scatter(const int* __restrict__ src, const int* __restrict__ dst,
                              const float* __restrict__ feat, const float* __restrict__ dinv,
                              float* __restrict__ agg) {
    const int gid   = blockIdx.x * 256 + threadIdx.x;
    const int slot  = gid >> 5;          // edge slot
    const int chunk = gid & 31;          // 32 chunks of 4 floats
    if (slot >= N_EDGES + N_NODES) return;
    int s, d;
    if (slot < N_EDGES) { s = src[slot]; d = dst[slot]; }
    else                { s = d = slot - N_EDGES; }
    const float di = dinv[s];
    const float4 f = *(const float4*)(feat + s * H_DIM + chunk * 4);
    float* ap = agg + d * H_DIM + chunk * 4;
    (void)unsafeAtomicAdd(ap + 0, f.x * di);
    (void)unsafeAtomicAdd(ap + 1, f.y * di);
    (void)unsafeAtomicAdd(ap + 2, f.z * di);
    (void)unsafeAtomicAdd(ap + 3, f.w * di);
}

// ---------------------------------------------------------------------------
// Update: feat -= agg*dinv ; agg = 0 ; acc_t += theta[t][k] * feat_new
// ---------------------------------------------------------------------------
__global__ void msgad_update(float* __restrict__ feat, float* __restrict__ agg,
                             const float* __restrict__ dinv,
                             float* __restrict__ acc0, float* __restrict__ acc1,
                             float* __restrict__ acc2, float* __restrict__ acc3,
                             float t0, float t1, float t2, float t3) {
    const int gid = blockIdx.x * 256 + threadIdx.x;   // < N*H
    const int n   = gid >> 7;
    const float di = dinv[n];
    const float f  = feat[gid] - agg[gid] * di;
    feat[gid] = f;
    agg[gid]  = 0.0f;
    acc0[gid] += t0 * f;
    acc1[gid] += t1 * f;
    acc2[gid] += t2 * f;
    acc3[gid] += t3 * f;
}

// ---------------------------------------------------------------------------
// Split f32 acc into f16 hi/lo pair (Markidis split) for WMMA.
// acc/hi/lo are flat over all 4 thetas (4*N*H elements).
// ---------------------------------------------------------------------------
__global__ void msgad_split(const float* __restrict__ acc,
                            _Float16* __restrict__ hi, _Float16* __restrict__ lo) {
    const int gid = blockIdx.x * 256 + threadIdx.x;   // < 4*N*H
    const float v = acc[gid];
    const _Float16 h = (_Float16)v;
    hi[gid] = h;
    lo[gid] = (_Float16)(v - (float)h);
}

// ---------------------------------------------------------------------------
// Gram kernel: out[t] = acc_t @ acc_t^T via split-f16 WMMA.
// Block = 128 threads (4 waves), each wave -> one 32x32 output tile (2x2 WMMA tiles).
// Grid (128, 128, 4): 64x64 tile per block.
// ---------------------------------------------------------------------------
__global__ __launch_bounds__(128)
void msgad_gram(const _Float16* __restrict__ Ahi, const _Float16* __restrict__ Alo,
                float* __restrict__ out) {
    const int t = blockIdx.z;
    const _Float16* __restrict__ Hh = Ahi + (size_t)t * NH;
    const _Float16* __restrict__ Lh = Alo + (size_t)t * NH;
    float* __restrict__ O = out + (size_t)t * N_NODES * N_NODES;

    const int wave = threadIdx.x >> 5;
    const int lane = threadIdx.x & 31;
    const int m0 = blockIdx.x * 64 + (wave >> 1) * 32;
    const int n0 = blockIdx.y * 64 + (wave & 1) * 32;

    // A fragment addressing (16x32 f16): lane L holds row m=L&15,
    // elems 0..7 -> K=kbA..kbA+7, elems 8..15 -> K=kbA+16..kbA+23, kbA=(L>>4)*8
    const int mA  = lane & 15;
    const int kbA = (lane >> 4) * 8;
    // B fragment (32x16 f16, == A^T): lane L holds col n=L&15,
    // elems 0..7 -> K=kbB..kbB+7, elems 8..15 -> K=kbB+16..kbB+23, kbB=(L>>4)*8
    const int nB  = lane & 15;
    const int kbB = (lane >> 4) * 8;

    v8f c[2][2] = {};

#pragma unroll
    for (int ks = 0; ks < 4; ++ks) {
        const int kofs = ks * 32;
        v16h ah[2], al[2], bh[2], bl[2];
#pragma unroll
        for (int ri = 0; ri < 2; ++ri) {
            const int base = (m0 + ri * 16 + mA) * H_DIM + kofs + kbA;
            v8h p0 = *(const v8h*)(Hh + base);
            v8h p1 = *(const v8h*)(Hh + base + 16);
            ah[ri] = __builtin_shufflevector(p0, p1, 0,1,2,3,4,5,6,7,8,9,10,11,12,13,14,15);
            v8h q0 = *(const v8h*)(Lh + base);
            v8h q1 = *(const v8h*)(Lh + base + 16);
            al[ri] = __builtin_shufflevector(q0, q1, 0,1,2,3,4,5,6,7,8,9,10,11,12,13,14,15);
        }
#pragma unroll
        for (int ci = 0; ci < 2; ++ci) {
            const int base = (n0 + ci * 16 + nB) * H_DIM + kofs + kbB;
            v8h p0 = *(const v8h*)(Hh + base);
            v8h p1 = *(const v8h*)(Hh + base + 16);
            bh[ci] = __builtin_shufflevector(p0, p1, 0,1,2,3,4,5,6,7,8,9,10,11,12,13,14,15);
            v8h q0 = *(const v8h*)(Lh + base);
            v8h q1 = *(const v8h*)(Lh + base + 16);
            bl[ci] = __builtin_shufflevector(q0, q1, 0,1,2,3,4,5,6,7,8,9,10,11,12,13,14,15);
        }
#pragma unroll
        for (int ri = 0; ri < 2; ++ri)
#pragma unroll
            for (int ci = 0; ci < 2; ++ci) {
                c[ri][ci] = __builtin_amdgcn_wmma_f32_16x16x32_f16(
                    false, ah[ri], false, bh[ci], (short)0, c[ri][ci], false, false);
                c[ri][ci] = __builtin_amdgcn_wmma_f32_16x16x32_f16(
                    false, ah[ri], false, bl[ci], (short)0, c[ri][ci], false, false);
                c[ri][ci] = __builtin_amdgcn_wmma_f32_16x16x32_f16(
                    false, al[ri], false, bh[ci], (short)0, c[ri][ci], false, false);
            }
    }

    // C/D layout: lane L -> col n=L&15; VGPR r -> row (L>=16 ? 8 : 0)+r
    const int nOut = lane & 15;
    const int mb   = (lane >> 4) * 8;
#pragma unroll
    for (int ri = 0; ri < 2; ++ri)
#pragma unroll
        for (int ci = 0; ci < 2; ++ci)
#pragma unroll
            for (int r = 0; r < 8; ++r)
                O[(size_t)(m0 + ri * 16 + mb + r) * N_NODES + (n0 + ci * 16 + nOut)]
                    = c[ri][ci][r];
}

// ---------------------------------------------------------------------------
// Launch
// ---------------------------------------------------------------------------
extern "C" void kernel_launch(void* const* d_in, const int* in_sizes, int n_in,
                              void* d_out, int out_size, void* d_ws, size_t ws_size,
                              hipStream_t stream) {
    (void)in_sizes; (void)n_in; (void)out_size; (void)ws_size;
    const float* X   = (const float*)d_in[0];
    const float* W   = (const float*)d_in[1];
    const float* b   = (const float*)d_in[2];
    const int*   src = (const int*)d_in[3];
    const int*   dst = (const int*)d_in[4];
    float* out = (float*)d_out;

    char* ws = (char*)d_ws;
    float*    feat = (float*)(ws);                         //  4 MB
    float*    agg  = (float*)(ws + ((size_t)4  << 20));    //  4 MB
    float*    acc  = (float*)(ws + ((size_t)8  << 20));    // 16 MB (4 x N*H)
    _Float16* hi   = (_Float16*)(ws + ((size_t)24 << 20)); //  8 MB
    _Float16* lo   = (_Float16*)(ws + ((size_t)32 << 20)); //  8 MB
    float*    deg  = (float*)(ws + ((size_t)40 << 20));    // 32 KB
    float*    dinv = deg + N_NODES;                        // 32 KB

    float* acc0 = acc;
    float* acc1 = acc + NH;
    float* acc2 = acc + 2 * NH;
    float* acc3 = acc + 3 * NH;

    msgad_gemm_leaky<<<N_NODES, 128, 0, stream>>>(X, W, b, feat, agg, acc0, acc1, acc2, acc3);
    msgad_deg_init<<<N_NODES / 256, 256, 0, stream>>>(deg);
    msgad_deg_count<<<N_EDGES / 256, 256, 0, stream>>>(src, deg);
    msgad_dinv<<<N_NODES / 256, 256, 0, stream>>>(deg, dinv);

    // theta[t][k] for k = 1..3 (theta from calc_thetas(4))
    const float TH[3][4] = {
        { -5.00f,  5.00f,  0.00f, 0.00f },   // k=1
        {  3.75f, -7.50f,  3.75f, 0.00f },   // k=2
        { -1.25f,  3.75f, -3.75f, 1.25f },   // k=3
    };
    const int scatterBlocks = ((N_EDGES + N_NODES) * 32) / 256;  // 33792
    for (int k = 0; k < 3; ++k) {
        msgad_scatter<<<scatterBlocks, 256, 0, stream>>>(src, dst, feat, dinv, agg);
        msgad_update<<<NH / 256, 256, 0, stream>>>(feat, agg, dinv,
                                                   acc0, acc1, acc2, acc3,
                                                   TH[k][0], TH[k][1], TH[k][2], TH[k][3]);
    }

    msgad_split<<<(4 * NH) / 256, 256, 0, stream>>>(acc, hi, lo);

    dim3 gg(N_NODES / 64, N_NODES / 64, 4);   // (128, 128, 4)
    msgad_gram<<<gg, 128, 0, stream>>>(hi, lo, out);
}
